// TreeNN_69861938037596
// MI455X (gfx1250) — matrix operand order repaired and checked by reading, not compiled, not run on hardware
//
#include <hip/hip_runtime.h>
#include <hip/hip_bf16.h>
#include <math.h>

// ---------------------------------------------------------------------------
// TreeNN forward for MI455X (gfx1250): all GEMMs via v_wmma_f32_16x16x32_f16.
// Round 2: vectorized LDS staging (float4 global loads, v4h/v8h LDS stores).
// ---------------------------------------------------------------------------

typedef __attribute__((ext_vector_type(4)))  _Float16 v4h;
typedef __attribute__((ext_vector_type(8)))  _Float16 v8h;
typedef __attribute__((ext_vector_type(16))) _Float16 v16h;
typedef __attribute__((ext_vector_type(8)))  float    v8f;

#define WMMA_F16(a, b, c) \
  __builtin_amdgcn_wmma_f32_16x16x32_f16(false, (a), false, (b), (short)0, (c), false, false)

// A fragment: 16x32 (MxK) 16-bit layout.
// lane L: m = L&15, kb = 8*(L>>4); halves 0..7 -> K=kb..kb+7, halves 8..15 -> K=16+kb..16+kb+7
__device__ __forceinline__ v16h load_fragA(const _Float16* base, int ld) {
  const int lane = threadIdx.x & 31;
  const int m  = lane & 15;
  const int kb = (lane >> 4) << 3;
  const _Float16* p = base + m * ld + kb;
  v8h lo = *(const v8h*)(p);
  v8h hi = *(const v8h*)(p + 16);
  v16h f;
#pragma unroll
  for (int i = 0; i < 8; ++i) { f[i] = lo[i]; f[i + 8] = hi[i]; }
  return f;
}

// B fragment: 32x16 (KxN) 16-bit layout, sourced from row-major Wt[N][K].
// lane L: n = L&15, kb = 8*(L>>4); halves 0..7 -> K=kb..kb+7, halves 8..15 -> K=16+kb..16+kb+7
__device__ __forceinline__ v16h load_fragB(const _Float16* base, int ld) {
  const int lane = threadIdx.x & 31;
  const int n  = lane & 15;
  const int kb = (lane >> 4) << 3;
  const _Float16* p = base + n * ld + kb;
  v8h lo = *(const v8h*)(p);
  v8h hi = *(const v8h*)(p + 16);
  v16h f;
#pragma unroll
  for (int i = 0; i < 8; ++i) { f[i] = lo[i]; f[i + 8] = hi[i]; }
  return f;
}

__device__ __forceinline__ v4h cvt4(float4 f) {
  v4h h;
  h[0] = (_Float16)f.x; h[1] = (_Float16)f.y;
  h[2] = (_Float16)f.z; h[3] = (_Float16)f.w;
  return h;
}

// ---------------------------------------------------------------------------
// Weight prep: fp32 W[K][N] -> f16 Wt[N][Kpad] (transposed, K zero-padded)
// ---------------------------------------------------------------------------
__global__ void wprep(const float* __restrict__ src, int K, int N, int Kpad,
                      _Float16* __restrict__ dst) {
  int i = blockIdx.x * blockDim.x + threadIdx.x;
  if (i >= N * Kpad) return;
  int n = i / Kpad, k = i - n * Kpad;
  float v = (k < K) ? src[(size_t)k * N + n] : 0.0f;
  dst[i] = (_Float16)v;
}

__global__ void zero_ker(float4* __restrict__ p, long long n4) {
  long long i = (long long)blockIdx.x * blockDim.x + threadIdx.x;
  if (i < n4) p[i] = make_float4(0.f, 0.f, 0.f, 0.f);
}

// ---------------------------------------------------------------------------
// Embedding kernel: feat[r][0:256] = [op_vec | c1 | c2 | bm] as f16.
//   rows r = b*63 + n, 64 rows per block, 256 threads (8 waves).
// ---------------------------------------------------------------------------
__global__ __launch_bounds__(256) void embed_kernel(
    const float* __restrict__ op, const float* __restrict__ extra,
    const float* __restrict__ cond1, const float* __restrict__ cond2,
    const float* __restrict__ bitmap, const float* __restrict__ has_cond,
    const _Float16* __restrict__ Wt_op,   const float* __restrict__ b_op,
    const _Float16* __restrict__ Wt_pred, const float* __restrict__ b_pred,
    const _Float16* __restrict__ Wt_bm,   const float* __restrict__ b_bm,
    _Float16* __restrict__ feat) {
  __shared__ _Float16 As[64 * 72];  // 64 rows x 64 cols, stride 72 halves
  const int R0 = blockIdx.x * 64;
  const int t = threadIdx.x;
  const int w = t >> 5;
  const int lane = t & 31;
  const v8f z8 = {0.f, 0.f, 0.f, 0.f, 0.f, 0.f, 0.f, 0.f};

  // ----- sections 0..2: K=64 GEMMs (op|extra, cond1, cond2) -----
  for (int sec = 0; sec < 3; ++sec) {
    // stage 64x64 fp32 -> f16 LDS, quad-vectorized (16 quads per row)
    for (int idx = t; idx < 64 * 16; idx += 256) {
      int row = idx >> 4, q = idx & 15;
      int col = q * 4;
      int r = R0 + row;
      float4 v4;
      if (sec == 0) {
        v4 = (col < 32) ? *(const float4*)(op + (size_t)r * 32 + col)
                        : *(const float4*)(extra + (size_t)r * 32 + (col - 32));
      } else if (sec == 1) {
        v4 = *(const float4*)(cond1 + (size_t)r * 64 + col);
      } else {
        v4 = *(const float4*)(cond2 + (size_t)r * 64 + col);
      }
      *(v4h*)(&As[row * 72 + col]) = cvt4(v4);
    }
    __syncthreads();
    const _Float16* Wt = (sec == 0) ? Wt_op : Wt_pred;
    const float* bias  = (sec == 0) ? b_op : b_pred;
    for (int tt = 0; tt < 2; ++tt) {
      int tile = w * 2 + tt;
      int mt = tile >> 2, nt = tile & 3;
      v8f acc = z8;
#pragma unroll
      for (int kk = 0; kk < 2; ++kk) {
        v16h a  = load_fragA(&As[mt * 16 * 72 + kk * 32], 72);
        v16h bf = load_fragB(Wt + (size_t)(nt * 16) * 64 + kk * 32, 64);
        acc = WMMA_F16(a, bf, acc);
      }
      int n = lane & 15, col = nt * 16 + n;
      float bb = bias[col];
#pragma unroll
      for (int rr = 0; rr < 8; ++rr) {
        int m = mt * 16 + rr + ((lane >> 4) << 3);
        float v = acc[rr] + bb;
        feat[(size_t)(R0 + m) * 256 + sec * 64 + col] = (_Float16)v;
      }
    }
    __syncthreads();
  }

  // ----- section 3: bitmap GEMM, K = 1000 padded to 1024 (16 chunks of 64) -----
  // 1000 = 250 quads: rows are 16B aligned, pad region is whole quads.
  v8f accb[2];
  accb[0] = z8; accb[1] = z8;
  for (int c = 0; c < 16; ++c) {
    for (int idx = t; idx < 64 * 16; idx += 256) {
      int row = idx >> 4, q = idx & 15;
      int gc = c * 64 + q * 4;
      int r = R0 + row;
      float4 v4 = (gc < 1000) ? *(const float4*)(bitmap + (size_t)r * 1000 + gc)
                              : make_float4(0.f, 0.f, 0.f, 0.f);
      *(v4h*)(&As[row * 72 + q * 4]) = cvt4(v4);
    }
    __syncthreads();
    for (int tt = 0; tt < 2; ++tt) {
      int tile = w * 2 + tt;
      int mt = tile >> 2, nt = tile & 3;
#pragma unroll
      for (int kk = 0; kk < 2; ++kk) {
        v16h a  = load_fragA(&As[mt * 16 * 72 + kk * 32], 72);
        v16h bf = load_fragB(Wt_bm + (size_t)(nt * 16) * 1024 + c * 64 + kk * 32, 1024);
        accb[tt] = WMMA_F16(a, bf, accb[tt]);
      }
    }
    __syncthreads();
  }
  for (int tt = 0; tt < 2; ++tt) {
    int tile = w * 2 + tt;
    int mt = tile >> 2, nt = tile & 3;
    int n = lane & 15, col = nt * 16 + n;
    float bb = b_bm[col];
#pragma unroll
    for (int rr = 0; rr < 8; ++rr) {
      int m = mt * 16 + rr + ((lane >> 4) << 3);
      int r = R0 + m;
      float v = (accb[tt][rr] + bb) * has_cond[r];
      feat[(size_t)r * 256 + 192 + col] = (_Float16)v;
    }
  }
}

// ---------------------------------------------------------------------------
// Tree level kernel: for each node at this level,
//   x = [feat(256) | rep(left,128) | rep(right,128)]  (512)
//   h = relu(x @ Wt_r1^T + b_r1)  (256)   -> LDS
//   r = relu(h @ Wt_r2^T + b_r2)  (128)   -> rep[node]
// rep indexed by global node id (0..126), node rows >=63 stay zero (leaves).
// ---------------------------------------------------------------------------
__global__ __launch_bounds__(256) void tree_kernel(
    int s, int lbits,
    const _Float16* __restrict__ feat,
    const _Float16* __restrict__ Wt_r1, const float* __restrict__ b_r1,
    const _Float16* __restrict__ Wt_r2, const float* __restrict__ b_r2,
    _Float16* __restrict__ rep) {
  __shared__ _Float16 Xs[64 * 136];  // 64 x 128 chunk of x, stride 136
  __shared__ _Float16 Hs[64 * 264];  // 64 x 256 h tile, stride 264
  const int R0 = blockIdx.x * 64;
  const int t = threadIdx.x;
  const int w = t >> 5;
  const int lane = t & 31;
  const int nlm1 = (1 << lbits) - 1;
  const v8f z8 = {0.f, 0.f, 0.f, 0.f, 0.f, 0.f, 0.f, 0.f};

  v8f acc[8];
#pragma unroll
  for (int i = 0; i < 8; ++i) acc[i] = z8;

  // ----- GEMM1: h(64x256) = x(64x512) @ W_r1, streamed over 4 K-chunks -----
  // Gather is 16B-vectorized: chunk boundaries (256, 384) are oct-aligned.
  for (int c = 0; c < 4; ++c) {
    for (int idx = t; idx < 64 * 16; idx += 256) {
      int row = idx >> 4, o = idx & 15;
      int gc = c * 128 + o * 8;
      int r = R0 + row;
      int b = r >> lbits;
      int node = s + (r & nlm1);
      v8h v;
      if (gc < 256)
        v = *(const v8h*)(feat + ((size_t)b * 63 + node) * 256 + gc);
      else if (gc < 384)
        v = *(const v8h*)(rep + ((size_t)b * 128 + (2 * node + 1)) * 128 + (gc - 256));
      else
        v = *(const v8h*)(rep + ((size_t)b * 128 + (2 * node + 2)) * 128 + (gc - 384));
      *(v8h*)(&Xs[row * 136 + o * 8]) = v;
    }
    __syncthreads();
    for (int tt = 0; tt < 8; ++tt) {
      int tile = w * 8 + tt;
      int mt = tile >> 4, nt = tile & 15;
#pragma unroll
      for (int kk = 0; kk < 4; ++kk) {
        v16h a  = load_fragA(&Xs[mt * 16 * 136 + kk * 32], 136);
        v16h bf = load_fragB(Wt_r1 + (size_t)(nt * 16) * 512 + c * 128 + kk * 32, 512);
        acc[tt] = WMMA_F16(a, bf, acc[tt]);
      }
    }
    __syncthreads();
  }
  // epilogue: relu -> Hs (f16)
  for (int tt = 0; tt < 8; ++tt) {
    int tile = w * 8 + tt;
    int mt = tile >> 4, nt = tile & 15;
    int n = lane & 15, col = nt * 16 + n;
    float bb = b_r1[col];
#pragma unroll
    for (int rr = 0; rr < 8; ++rr) {
      int m = mt * 16 + rr + ((lane >> 4) << 3);
      float v = acc[tt][rr] + bb;
      Hs[m * 264 + col] = (_Float16)(v > 0.f ? v : 0.f);
    }
  }
  __syncthreads();

  // ----- GEMM2: r(64x128) = h(64x256) @ W_r2, relu -> rep -----
  for (int tt = 0; tt < 4; ++tt) {
    int tile = w * 4 + tt;
    int mt = tile >> 3, nt = tile & 7;
    v8f a2 = z8;
#pragma unroll
    for (int kk = 0; kk < 8; ++kk) {
      v16h a  = load_fragA(&Hs[mt * 16 * 264 + kk * 32], 264);
      v16h bf = load_fragB(Wt_r2 + (size_t)(nt * 16) * 256 + kk * 32, 256);
      a2 = WMMA_F16(a, bf, a2);
    }
    int n = lane & 15, col = nt * 16 + n;
    float bb = b_r2[col];
#pragma unroll
    for (int rr = 0; rr < 8; ++rr) {
      int m = mt * 16 + rr + ((lane >> 4) << 3);
      int r = R0 + m;
      int b = r >> lbits;
      int node = s + (r & nlm1);
      float v = a2[rr] + bb;
      rep[((size_t)b * 128 + node) * 128 + col] = (_Float16)(v > 0.f ? v : 0.f);
    }
  }
}

// ---------------------------------------------------------------------------
// Heads: cost/card MLPs on root rep. 1 block per batch row, 128 threads
// (threads 0..63 = cost head, 64..127 = card head).
// ---------------------------------------------------------------------------
__global__ __launch_bounds__(128) void head_kernel(
    const _Float16* __restrict__ rep,
    const float* __restrict__ Wc1, const float* __restrict__ bc1,
    const float* __restrict__ Wc2, const float* __restrict__ bc2,
    const float* __restrict__ Wc3, const float* __restrict__ bc3,
    const float* __restrict__ Wd1, const float* __restrict__ bd1,
    const float* __restrict__ Wd2, const float* __restrict__ bd2,
    const float* __restrict__ Wd3, const float* __restrict__ bd3,
    float* __restrict__ out) {
  __shared__ float root[128];
  __shared__ float h1[128];
  __shared__ float h2[128];
  __shared__ float red[128];
  const int b = blockIdx.x;
  const int t = threadIdx.x;
  const int head = t >> 6;
  const int j = t & 63;

  root[t] = (float)rep[((size_t)b * 128 + 0) * 128 + t];
  __syncthreads();

  const float* W1 = head ? Wd1 : Wc1;
  const float* B1 = head ? bd1 : bc1;
  const float* W2 = head ? Wd2 : Wc2;
  const float* B2 = head ? bd2 : bc2;
  const float* W3 = head ? Wd3 : Wc3;
  const float* B3 = head ? bd3 : bc3;

  float s1 = B1[j];
  for (int k = 0; k < 128; ++k) s1 += root[k] * W1[k * 64 + j];
  h1[t] = s1 > 0.f ? s1 : 0.f;
  __syncthreads();

  float s2 = B2[j];
  for (int k = 0; k < 64; ++k) s2 += h1[head * 64 + k] * W2[k * 64 + j];
  h2[t] = s2 > 0.f ? s2 : 0.f;
  __syncthreads();

  red[t] = h2[t] * W3[j];
  __syncthreads();
  for (int off = 32; off > 0; off >>= 1) {
    if (j < off) red[t] += red[t + off];
    __syncthreads();
  }
  if (j == 0) {
    float zv = red[t] + B3[0];
    out[head * 2048 + b] = 1.0f / (1.0f + expf(-zv));
  }
}

// ---------------------------------------------------------------------------
// Host launch
// ---------------------------------------------------------------------------
extern "C" void kernel_launch(void* const* d_in, const int* in_sizes, int n_in,
                              void* d_out, int out_size, void* d_ws, size_t ws_size,
                              hipStream_t stream) {
  (void)in_sizes; (void)n_in; (void)out_size; (void)ws_size;
  const float* op       = (const float*)d_in[0];
  const float* extra    = (const float*)d_in[1];
  const float* cond1    = (const float*)d_in[2];
  const float* cond2    = (const float*)d_in[3];
  const float* bitmap   = (const float*)d_in[4];
  const float* has_cond = (const float*)d_in[5];
  const float* W_op = (const float*)d_in[6];   const float* b_op = (const float*)d_in[7];
  const float* W_pr = (const float*)d_in[8];   const float* b_pr = (const float*)d_in[9];
  const float* W_bm = (const float*)d_in[10];  const float* b_bm = (const float*)d_in[11];
  const float* W_r1 = (const float*)d_in[12];  const float* b_r1 = (const float*)d_in[13];
  const float* W_r2 = (const float*)d_in[14];  const float* b_r2 = (const float*)d_in[15];
  const float* W_c1 = (const float*)d_in[16];  const float* b_c1 = (const float*)d_in[17];
  const float* W_c2 = (const float*)d_in[18];  const float* b_c2 = (const float*)d_in[19];
  const float* W_c3 = (const float*)d_in[20];  const float* b_c3 = (const float*)d_in[21];
  const float* W_d1 = (const float*)d_in[22];  const float* b_d1 = (const float*)d_in[23];
  const float* W_d2 = (const float*)d_in[24];  const float* b_d2 = (const float*)d_in[25];
  const float* W_d3 = (const float*)d_in[26];  const float* b_d3 = (const float*)d_in[27];
  float* out = (float*)d_out;

  // ---- workspace partition (256B aligned) ----
  char* base = (char*)d_ws;
  size_t off = 0;
  auto take = [&](size_t bytes) -> char* {
    char* q = base + off;
    off += bytes;
    off = (off + 255) & ~(size_t)255;
    return q;
  };
  _Float16* Wt_op = (_Float16*)take((size_t)64 * 64 * 2);
  _Float16* Wt_pr = (_Float16*)take((size_t)64 * 64 * 2);
  _Float16* Wt_bm = (_Float16*)take((size_t)64 * 1024 * 2);
  _Float16* Wt_r1 = (_Float16*)take((size_t)256 * 512 * 2);
  _Float16* Wt_r2 = (_Float16*)take((size_t)128 * 256 * 2);
  _Float16* feat  = (_Float16*)take((size_t)129024 * 256 * 2);     // [B*63][256]
  _Float16* rep   = (_Float16*)take((size_t)2048 * 128 * 128 * 2); // [B][128 nodes][128]

  // ---- weight prep (transpose + f16 convert, K zero-padded) ----
  wprep<<<(64 * 64 + 255) / 256, 256, 0, stream>>>(W_op, 64, 64, 64, Wt_op);
  wprep<<<(64 * 64 + 255) / 256, 256, 0, stream>>>(W_pr, 64, 64, 64, Wt_pr);
  wprep<<<(64 * 1024 + 255) / 256, 256, 0, stream>>>(W_bm, 1000, 64, 1024, Wt_bm);
  wprep<<<(256 * 512 + 255) / 256, 256, 0, stream>>>(W_r1, 512, 256, 512, Wt_r1);
  wprep<<<(128 * 256 + 255) / 256, 256, 0, stream>>>(W_r2, 256, 128, 256, Wt_r2);

  // ---- zero rep (leaf children must read zeros; deterministic each call) ----
  long long n4 = (long long)2048 * 128 * 128 * 2 / 16;
  zero_ker<<<(unsigned)((n4 + 255) / 256), 256, 0, stream>>>((float4*)rep, n4);

  // ---- embeddings: 129024 rows / 64 per block ----
  embed_kernel<<<2016, 256, 0, stream>>>(op, extra, cond1, cond2, bitmap, has_cond,
                                         Wt_op, b_op, Wt_pr, b_pr, Wt_bm, b_bm, feat);

  // ---- tree levels, bottom-up ----
  for (int l = 5; l >= 0; --l) {
    int nl = 1 << l;
    int sidx = nl - 1;
    int blocks = (2048 * nl) / 64;
    tree_kernel<<<blocks, 256, 0, stream>>>(sidx, l, feat, Wt_r1, b_r1, Wt_r2, b_r2, rep);
  }

  // ---- heads ----
  head_kernel<<<2048, 128, 0, stream>>>(rep, W_c1, b_c1, W_c2, b_c2, W_c3, b_c3,
                                        W_d1, b_d1, W_d2, b_d2, W_d3, b_d3, out);
}
